// BidirectionalLSTM_19138374271110
// MI455X (gfx1250) — matrix operand order repaired and checked by reading, not compiled, not run on hardware
//
#include <hip/hip_runtime.h>
#include <hip/hip_bf16.h>

// ---------------------------------------------------------------------------
// Bidirectional 2-layer LSTM for MI455X (gfx1250), wave32 + WMMA bf16 + TDM.
//   B=32, T=2048, F=H=128, gates 4H=512.
// Phase 1: input projection GEMMs (parallel, WMMA bf16, f32 accumulate)
// Phase 2: persistent-workgroup recurrent scan (1 WGP per direction):
//   - Wh held as WMMA B-fragments in registers for all 2048 steps
//   - xp[t] slab (32x512 f32, 64KB) staged into LDS by the Tensor Data Mover,
//     DMA overlapped with the elementwise cell-update phase
//   - h/c/gates in LDS, barriers between WMMA and cell-update phases
// ---------------------------------------------------------------------------

#define HH   128
#define BB   32
#define TT   2048
#define G4   512   // 4*H

typedef __attribute__((ext_vector_type(16))) __bf16 v16bf;
typedef __attribute__((ext_vector_type(8)))  __bf16 v8bf;
typedef __attribute__((ext_vector_type(8)))  float  v8f;
typedef __attribute__((ext_vector_type(4))) unsigned int u32x4;
typedef __attribute__((ext_vector_type(8))) int          i32x8;
typedef __attribute__((ext_vector_type(4))) int          i32x4;

__device__ __forceinline__ float sigmoidf_(float x) {
    return 1.0f / (1.0f + __expf(-x));
}

// ---------------------------------------------------------------------------
// TDM: DMA one (32 x 512) f32 tile from global into LDS at byte offset
// lds_off.  D# per CDNA5 ISA (2-D tensor, groups 2/3
// zero; 6-arg clang-23 builtin form).  Tracked by TENSORcnt.
// ---------------------------------------------------------------------------
__device__ __forceinline__ void tdm_load_slab(const float* gsrc, unsigned lds_off) {
    const unsigned long long ga = (unsigned long long)(uintptr_t)gsrc;
    u32x4 g0;
    g0[0] = 1u;                                   // count=1 (valid), user mode
    g0[1] = lds_off;                              // lds_addr (bytes)
    g0[2] = (unsigned)(ga & 0xFFFFFFFFu);         // global_addr[31:0]
    g0[3] = (unsigned)(ga >> 32) | (2u << 30);    // global_addr[56:32] | type=2
    i32x8 g1;
    g1[0] = (int)(2u << 16);                      // data_size = 4 bytes
    g1[1] = (int)((unsigned)(G4 & 0xFFFF) << 16); // tensor_dim0[15:0] = 512
    g1[2] = (int)((unsigned)(BB & 0xFFFF) << 16); // tensor_dim0 hi=0 | tensor_dim1 lo=32
    g1[3] = (int)((unsigned)G4 << 16);            // tensor_dim1 hi=0 | tile_dim0=512
    g1[4] = (int)BB;                              // tile_dim1=32, tile_dim2=0
    g1[5] = (int)G4;                              // tensor_dim0_stride = 512
    g1[6] = 0;
    g1[7] = 0;
    const i32x4 z4 = {0, 0, 0, 0};
    const i32x8 z8 = {0, 0, 0, 0, 0, 0, 0, 0};
    __builtin_amdgcn_tensor_load_to_lds(g0, g1, z4, z4, z8, 0);
}

// ---------------------------------------------------------------------------
// Input projection: XP[t][b][g] = bias[g] + sum_f X[b][t][f] * W[f][g]
//   X : (B, T, Fin) f32 | W : (Fin, 512) f32 | XP : (T, B, 512) f32
// One 16x16 output tile per wave, K consumed 32 at a time via WMMA bf16.
// ---------------------------------------------------------------------------
__global__ __launch_bounds__(256) void lstm_proj(
    const float* __restrict__ X,
    const float* __restrict__ W,
    const float* __restrict__ bias,
    float* __restrict__ XP,
    int Fin)
{
    const int lane  = threadIdx.x & 31;
    const int wid   = threadIdx.x >> 5;
    const int gw    = blockIdx.x * 8 + wid;   // global wave id
    const int ntile = gw & 31;                // 32 N-tiles of 16
    const int mtile = gw >> 5;                // (T*B)/16 M-tiles
    const int t     = mtile >> 1;             // B=32 -> 2 M-tiles per t
    const int bh    = mtile & 1;
    const int r     = lane & 15;
    const int kh    = lane >> 4;              // K-half selector
    const int n     = ntile * 16 + r;

    v8f acc;
    const float bv = bias[n];
#pragma unroll
    for (int v = 0; v < 8; ++v) acc[v] = bv;

    const int    brow = bh * 16 + r;
    const float* arow = X + ((size_t)brow * TT + t) * Fin;
    const int    nks  = Fin >> 5;

    for (int ks = 0; ks < nks; ++ks) {
        // A fragment: 16-bit A 16x32 layout — lane kh half, pairs along K
        v16bf a;
#pragma unroll
        for (int j = 0; j < 8; ++j) {
            const int off = ks * 32 + kh * 8 + ((j < 4) ? 2 * j : 8 + 2 * j);
            const float2 p = *(const float2*)(arow + off);
            a[2 * j]     = (__bf16)p.x;
            a[2 * j + 1] = (__bf16)p.y;
        }
        // B fragment: lanes 0-15 K=0..7,16..23, lanes 16-31 K=8..15,24..31, N = lane%16
        v16bf w;
#pragma unroll
        for (int v = 0; v < 8; ++v) {
            const int k = ks * 32 + kh * 8 + 2 * v + ((v >= 4) ? 8 : 0);
            w[2 * v]     = (__bf16)W[(size_t)k * G4 + n];
            w[2 * v + 1] = (__bf16)W[(size_t)(k + 1) * G4 + n];
        }
        acc = __builtin_amdgcn_wmma_f32_16x16x32_bf16(
                  false, a, false, w, (short)0, acc, false, false);
    }

#pragma unroll
    for (int v = 0; v < 8; ++v) {
        const int row = bh * 16 + kh * 8 + v;
        XP[((size_t)t * BB + row) * G4 + n] = acc[v];
    }
}

// ---------------------------------------------------------------------------
// Recurrent scan. grid = 2 (blockIdx.x = direction), block = 1024 (32 waves).
// Wave w owns gate columns [16w, 16w+16) and both 16-row M-tiles, so the Wh
// B-fragments (4 K-steps, 32 VGPRs) stay in registers for the whole scan.
// ---------------------------------------------------------------------------
__global__ __launch_bounds__(1024) void lstm_scan(
    const float* __restrict__ xp_f, const float* __restrict__ xp_b,
    const float* __restrict__ Wh_f, const float* __restrict__ Wh_b,
    float* __restrict__ out,
    float* __restrict__ cfin)
{
    const int dir = blockIdx.x;                 // 0 = forward, 1 = backward
    const float* __restrict__ xp = dir ? xp_b : xp_f;
    const float* __restrict__ Wh = dir ? Wh_b : Wh_f;

    __shared__ float  xbuf[BB * G4];            // 64 KB TDM-staged xp[t]
    __shared__ float  gbuf[BB * G4];            // 64 KB raw gates
    __shared__ float  cbuf[BB * HH];            // 16 KB cell state
    __shared__ __bf16 hbuf[BB * HH];            // 8 KB hidden, row-major (b,k)

    const int tid  = threadIdx.x;
    const int lane = tid & 31;
    const int wave = tid >> 5;                  // N-tile id
    const int r    = lane & 15;
    const int kh   = lane >> 4;
    const int n    = wave * 16 + r;
    const unsigned xbuf_lds = (unsigned)(uintptr_t)&xbuf[0]; // LDS byte offset

    // zero-init carry
    for (int i = tid; i < BB * HH; i += 1024) {
        cbuf[i] = 0.0f;
        hbuf[i] = (__bf16)0.0f;
    }

    // Persistent Wh B-fragments (bf16), 4 K-steps covering K=128
    v16bf bw[4];
#pragma unroll
    for (int ks = 0; ks < 4; ++ks) {
#pragma unroll
        for (int v = 0; v < 8; ++v) {
            const int k = ks * 32 + kh * 8 + 2 * v + ((v >= 4) ? 8 : 0);
            bw[ks][2 * v]     = (__bf16)Wh[(size_t)k * G4 + n];
            bw[ks][2 * v + 1] = (__bf16)Wh[(size_t)(k + 1) * G4 + n];
        }
    }

    // Stage first xp slab via TDM (wave 0 issues; barrier publishes)
    const int t0 = dir ? (TT - 1) : 0;
    if (wave == 0) {
        tdm_load_slab(xp + (size_t)t0 * BB * G4, xbuf_lds);
        __builtin_amdgcn_s_wait_tensorcnt(0);
    }
    __syncthreads();

    for (int step = 0; step < TT; ++step) {
        const int t = dir ? (TT - 1 - step) : step;

        // ----- gate GEMM: g = xp[t] + h @ Wh  (xp[t] in LDS xbuf) -----
#pragma unroll
        for (int m = 0; m < 2; ++m) {
            v8f acc;
#pragma unroll
            for (int v = 0; v < 8; ++v) {
                const int row = m * 16 + kh * 8 + v;     // C/D layout M-row
                acc[v] = xbuf[row * G4 + n];
            }
#pragma unroll
            for (int ks = 0; ks < 4; ++ks) {
                // A fragment: two ds_load_b128 from row-major bf16 hbuf
                const __bf16* ap = &hbuf[(m * 16 + r) * HH + ks * 32 + kh * 8];
                const v8bf lo = *(const v8bf*)ap;          // K+0..7
                const v8bf hi = *(const v8bf*)(ap + 16);   // K+16..23
                v16bf a;
#pragma unroll
                for (int e = 0; e < 8; ++e) { a[e] = lo[e]; a[8 + e] = hi[e]; }
                acc = __builtin_amdgcn_wmma_f32_16x16x32_bf16(
                          false, a, false, bw[ks], (short)0, acc, false, false);
            }
#pragma unroll
            for (int v = 0; v < 8; ++v) {
                const int row = m * 16 + kh * 8 + v;
                gbuf[row * G4 + n] = acc[v];
            }
        }
        __syncthreads();   // all waves done reading xbuf; gbuf published

        // Kick DMA for the next slab; it overlaps the cell-update phase.
        const int tn = dir ? (t - 1) : (t + 1);
        if (wave == 0 && tn >= 0 && tn < TT)
            tdm_load_slab(xp + (size_t)tn * BB * G4, xbuf_lds);

        // Warm L2 two steps ahead (64 B per thread covers the 64 KB slab)
        const int tp = dir ? (t - 2) : (t + 2);
        if (tp >= 0 && tp < TT)
            __builtin_prefetch(xp + (size_t)tp * BB * G4 + (tid << 4), 0, 1);

        // ----- cell update: 4 elements per thread (32*128 = 4096) -----
#pragma unroll
        for (int e = 0; e < 4; ++e) {
            const int idx = e * 1024 + tid;
            const int b = idx >> 7;
            const int j = idx & 127;
            const float gi = gbuf[b * G4 + j];
            const float gf = gbuf[b * G4 + 128 + j];
            const float gg = gbuf[b * G4 + 256 + j];
            const float go = gbuf[b * G4 + 384 + j];
            const float cn = sigmoidf_(gf) * cbuf[idx] + sigmoidf_(gi) * tanhf(gg);
            const float hn = sigmoidf_(go) * tanhf(cn);
            cbuf[idx] = cn;
            hbuf[idx] = (__bf16)hn;
            out[((size_t)b * TT + t) * 256 + dir * 128 + j] = hn;
        }

        if (wave == 0) __builtin_amdgcn_s_wait_tensorcnt(0);
        __syncthreads();   // next slab resident; hbuf/cbuf published
    }

    if (cfin) {
        for (int idx = tid; idx < BB * HH; idx += 1024) {
            const int b = idx >> 7;
            const int j = idx & 127;
            cfin[b * 256 + dir * 128 + j] = cbuf[idx];
        }
    }
}

// ---------------------------------------------------------------------------
// Inputs (setup_inputs dict order):
//  0 x | 1 Wx_f0 | 2 Wh_f0 | 3 Wx_b0 | 4 Wh_b0 | 5 Wx_f1 | 6 Wh_f1
//  7 Wx_b1 | 8 Wh_b1 | 9 b_f0 | 10 b_b0 | 11 b_f1 | 12 b_b1
// d_out: x_final (B,T,256) followed by c_final (B,256) — both f32.
// ---------------------------------------------------------------------------
extern "C" void kernel_launch(void* const* d_in, const int* in_sizes, int n_in,
                              void* d_out, int out_size, void* d_ws, size_t ws_size,
                              hipStream_t stream)
{
    const float* x     = (const float*)d_in[0];
    const float* Wx_f0 = (const float*)d_in[1];
    const float* Wh_f0 = (const float*)d_in[2];
    const float* Wx_b0 = (const float*)d_in[3];
    const float* Wh_b0 = (const float*)d_in[4];
    const float* Wx_f1 = (const float*)d_in[5];
    const float* Wh_f1 = (const float*)d_in[6];
    const float* Wx_b1 = (const float*)d_in[7];
    const float* Wh_b1 = (const float*)d_in[8];
    const float* b_f0  = (const float*)d_in[9];
    const float* b_b0  = (const float*)d_in[10];
    const float* b_f1  = (const float*)d_in[11];
    const float* b_b1  = (const float*)d_in[12];

    const size_t NP = (size_t)TT * BB * G4;        // 33.5M floats per xp buf
    float* ws   = (float*)d_ws;
    float* xp_f = ws;
    float* xp_b = ws + NP;
    float* x1   = ws + 2 * NP;                     // layer-0 output (B,T,256)

    float* outx = (float*)d_out;                   // (B,T,256)
    float* outc = outx + (size_t)BB * TT * 256;    // (B,256)

    // (T*B/16) M-tiles * 32 N-tiles / 8 waves per block = 16384 blocks
    const dim3 pg(16384), pb(256);
    const dim3 sg(2), sb(1024);

    // Layer 0
    lstm_proj<<<pg, pb, 0, stream>>>(x, Wx_f0, b_f0, xp_f, 128);
    lstm_proj<<<pg, pb, 0, stream>>>(x, Wx_b0, b_b0, xp_b, 128);
    lstm_scan<<<sg, sb, 0, stream>>>(xp_f, xp_b, Wh_f0, Wh_b0, x1, nullptr);

    // Layer 1 (input = concat(out_f0, out_b0) : (B,T,256))
    lstm_proj<<<pg, pb, 0, stream>>>(x1, Wx_f1, b_f1, xp_f, 256);
    lstm_proj<<<pg, pb, 0, stream>>>(x1, Wx_b1, b_b1, xp_b, 256);
    lstm_scan<<<sg, sb, 0, stream>>>(xp_f, xp_b, Wh_f1, Wh_b1, outx, outc);
}
